// DecoderLayer_17308718203506
// MI455X (gfx1250) — matrix operand order repaired and checked by reading, not compiled, not run on hardware
//
#include <hip/hip_runtime.h>
#include <cstdint>
#include <cstddef>

// ---------------------------------------------------------------------------
// MI455X (gfx1250) transformer decoder layer, bf16 WMMA + fp32 accumulation.
// All GEMM B operands pre-transposed (N x K) -> LDS staging is pure b128 copy,
// via gfx1250 async global->LDS; 2-stage pipelined main loop.
// ---------------------------------------------------------------------------

typedef __bf16 bf16;
typedef __attribute__((ext_vector_type(16))) __bf16 v16bf;
typedef __attribute__((ext_vector_type(8)))  float  v8f;
typedef int v4i_vs __attribute__((vector_size(16)));

#define DMODEL 1024
#define NHEADS 16
#define DFF    4096
#define BATCH  2
#define SEQ    2048
#define NEGV   (-1e9f)

#ifndef __has_builtin
#define __has_builtin(x) 0
#endif

#if defined(__HIP_DEVICE_COMPILE__) && \
    __has_builtin(__builtin_amdgcn_global_load_async_to_lds_b128)
#define ASYNC_LDS 1
#else
#define ASYNC_LDS 0
#endif

__device__ __forceinline__ bf16 f2bf(float f) { return (bf16)f; }

// 16-byte global -> LDS copy; async (ASYNCcnt-tracked) when available.
// Builtin signature (from probe): (v4i AS1*, v4i AS3*, imm offset, imm cpol).
__device__ __forceinline__ void copy16_g2l(const bf16* g, bf16* l) {
#if ASYNC_LDS
    __builtin_amdgcn_global_load_async_to_lds_b128(
        (__attribute__((address_space(1))) v4i_vs*)(g),
        (__attribute__((address_space(3))) v4i_vs*)(l), 0, 0);
#else
    *reinterpret_cast<float4*>(l) = *reinterpret_cast<const float4*>(g);
#endif
}

__device__ __forceinline__ void wait_async_all() {
#if ASYNC_LDS
#if __has_builtin(__builtin_amdgcn_s_wait_asynccnt)
    __builtin_amdgcn_s_wait_asynccnt(0);
#else
    asm volatile("s_wait_asynccnt 0x0" ::: "memory");
#endif
#endif
}

union AF { v16bf v; float4 q[2]; };

constexpr int BM = 128, BN = 64, BK = 32;
constexpr int SA = 40, SB = 40;   // padded LDS row strides (bf16 elems)

// ---------------------------------------------------------------------------
// Tiled WMMA GEMM:  C[M,N] = A[M,K] @ B^T  with B supplied as N x K row-major.
//   A_F32    : A is fp32, converted to bf16 while staging to LDS
//   MASK     : 0 none, 1 causal (k>q -> NEG), 2 padding mask per (b2, key)
//   OUT_TRANS: store C transposed (C[n][m]), contiguous b128 bf16 stores
//   HEAD     : 0 plain, 1 score GEMM per head-slot z, 2 attn@V GEMM per slot z
// 256 threads (8 waves); block tile 128x64; wave tile 32x32 (2x2 WMMA 16x16x32).
// Double-buffered LDS, software-pipelined (stage k+1 while computing k).
// ---------------------------------------------------------------------------
template<int A_F32, int BIAS, int RELU, int MASK, int OUT_BF16, int OUT_TRANS, int HEAD>
__global__ __launch_bounds__(256) void gemm_wmma(
    const void* __restrict__ Ap, int lda,
    const bf16* __restrict__ Bp, int ldb,
    void*       __restrict__ Cp, int ldc,
    const float* __restrict__ bias,
    const float* __restrict__ maskp,
    float scale, int K)
{
    __shared__ bf16 Alds[2][BM * SA];
    __shared__ bf16 Blds[2][BN * SB];

    const int tid = threadIdx.x;
    const int bm  = blockIdx.y * BM;
    const int bn  = blockIdx.x * BN;
    const int z   = blockIdx.z;

    // Per-head offsets (handles _split_heads reinterpret quirk:
    // output slot z=b2*16+h2 sources from (b,h)=(z&1, z>>1)).
    size_t aOff = 0, bOff = 0, cOff = 0;
    if (HEAD == 1) {
        int sb = z & 1, sh = z >> 1;
        aOff = (size_t)sb * SEQ * lda + (size_t)sh * 64;
        bOff = (size_t)sb * SEQ * ldb + (size_t)sh * 64;
        cOff = (size_t)z * SEQ * (size_t)ldc;
    }
    if (HEAD == 2) {   // A = softmaxed w (f32); B = V^T (DMODEL x B*S)
        int sb = z & 1, sh = z >> 1;
        aOff = (size_t)z * SEQ * (size_t)lda;
        bOff = (size_t)sh * 64 * (size_t)ldb + (size_t)sb * SEQ;
        cOff = (size_t)z * SEQ * (size_t)ldc;
    }

    const float* Afp = (const float*)Ap + aOff;
    const bf16*  Abp = (const bf16*)Ap + aOff;
    const bf16*  Bb  = Bp + bOff;

    const int lane = tid & 31, wv = tid >> 5;
    const int wm = wv >> 1, wn = wv & 1;     // 4x2 wave grid
    const int l15 = lane & 15, hh = lane >> 4;

    float4 aReg[4];   // register prefetch for the A_F32 path

    auto stage_issue = [&](int kk, int buf) {
        if (A_F32) {
            #pragma unroll
            for (int it = 0; it < 4; ++it) {
                int e = (tid + it * 256) * 4;
                int r = e >> 5, c = e & 31;
                aReg[it] = *reinterpret_cast<const float4*>(
                    Afp + (size_t)(bm + r) * lda + kk + c);
            }
        } else {
            #pragma unroll
            for (int it = 0; it < 2; ++it) {
                int e = (tid + it * 256) * 8;
                int r = e >> 5, c = e & 31;
                copy16_g2l(Abp + (size_t)(bm + r) * lda + kk + c,
                           &Alds[buf][r * SA + c]);
            }
        }
        int e = tid * 8; int n = e >> 5, c = e & 31;
        copy16_g2l(Bb + (size_t)(bn + n) * ldb + kk + c,
                   &Blds[buf][n * SB + c]);
    };
    auto stage_commit = [&](int buf) {
        if (A_F32) {
            #pragma unroll
            for (int it = 0; it < 4; ++it) {
                int e = (tid + it * 256) * 4;
                int r = e >> 5, c = e & 31;
                union { bf16 h[4]; uint2 u; } pk;
                pk.h[0] = f2bf(aReg[it].x); pk.h[1] = f2bf(aReg[it].y);
                pk.h[2] = f2bf(aReg[it].z); pk.h[3] = f2bf(aReg[it].w);
                *reinterpret_cast<uint2*>(&Alds[buf][r * SA + c]) = pk.u;
            }
        }
    };

    v8f acc[2][2] = {};
    const int nk = K / BK;

    stage_issue(0, 0);
    stage_commit(0);
    wait_async_all();
    __syncthreads();

    for (int k = 0; k < nk; ++k) {
        const int cur = k & 1, nxt = cur ^ 1;
        if (k + 1 < nk) stage_issue((k + 1) * BK, nxt);

        // fragments (ISA 16-bit A 16x32 / B 32x16 VGPR layouts)
        AF a0, a1, b0, b1;
        const bf16* ar0 = &Alds[cur][(wm * 32 +      l15) * SA];
        const bf16* ar1 = &Alds[cur][(wm * 32 + 16 + l15) * SA];
        a0.q[0] = *reinterpret_cast<const float4*>(ar0 + hh * 8);
        a0.q[1] = *reinterpret_cast<const float4*>(ar0 + 16 + hh * 8);
        a1.q[0] = *reinterpret_cast<const float4*>(ar1 + hh * 8);
        a1.q[1] = *reinterpret_cast<const float4*>(ar1 + 16 + hh * 8);
        const bf16* br0 = &Blds[cur][(wn * 32 +      l15) * SB];
        const bf16* br1 = &Blds[cur][(wn * 32 + 16 + l15) * SB];
        b0.q[0] = *reinterpret_cast<const float4*>(br0 + hh * 8);
        b0.q[1] = *reinterpret_cast<const float4*>(br0 + 16 + hh * 8);
        b1.q[0] = *reinterpret_cast<const float4*>(br1 + hh * 8);
        b1.q[1] = *reinterpret_cast<const float4*>(br1 + 16 + hh * 8);

        acc[0][0] = __builtin_amdgcn_wmma_f32_16x16x32_bf16(
            false, a0.v, false, b0.v, (short)0, acc[0][0], false, false);
        acc[0][1] = __builtin_amdgcn_wmma_f32_16x16x32_bf16(
            false, a0.v, false, b1.v, (short)0, acc[0][1], false, false);
        acc[1][0] = __builtin_amdgcn_wmma_f32_16x16x32_bf16(
            false, a1.v, false, b0.v, (short)0, acc[1][0], false, false);
        acc[1][1] = __builtin_amdgcn_wmma_f32_16x16x32_bf16(
            false, a1.v, false, b1.v, (short)0, acc[1][1], false, false);

        if (k + 1 < nk) stage_commit(nxt);
        wait_async_all();
        __syncthreads();
    }

    // ---- epilogue (C/D layout: lane -> N=l15, VGPR v -> M = 8*hh + v) ----
    float* Cf = (float*)Cp + cOff;
    bf16*  Cb = (bf16*)Cp + cOff;
    #pragma unroll
    for (int ti = 0; ti < 2; ++ti) {
        #pragma unroll
        for (int tj = 0; tj < 2; ++tj) {
            int gn  = bn + wn * 32 + tj * 16 + l15;
            int gmB = bm + wm * 32 + ti * 16 + hh * 8;
            if (OUT_BF16 && OUT_TRANS) {
                union { bf16 h[8]; float4 f; } pk;
                #pragma unroll
                for (int v2 = 0; v2 < 8; ++v2) {
                    float val = acc[ti][tj][v2];
                    if (BIAS) val += bias[gn];
                    if (RELU) val = fmaxf(val, 0.f);
                    pk.h[v2] = f2bf(val);
                }
                *reinterpret_cast<float4*>(&Cb[(size_t)gn * ldc + gmB]) = pk.f;
            } else {
                #pragma unroll
                for (int v2 = 0; v2 < 8; ++v2) {
                    int gm = gmB + v2;
                    float val = acc[ti][tj][v2];
                    if (BIAS) val += bias[gn];
                    if (MASK == 1) val = val * scale + (gn > gm ? NEGV : 0.f);
                    if (MASK == 2) val = val * scale +
                                         maskp[(size_t)(z >> 4) * SEQ + gn] * NEGV;
                    if (RELU) val = fmaxf(val, 0.f);
                    if (OUT_BF16) Cb[(size_t)gm * ldc + gn] = f2bf(val);
                    else          Cf[(size_t)gm * ldc + gn] = val;
                }
            }
        }
    }
}

// ---------------------------------------------------------------------------
// Row softmax over length-2048 rows, in place (one 256-thread block per row).
// ---------------------------------------------------------------------------
__global__ __launch_bounds__(256) void softmax2048(float* __restrict__ w)
{
    __shared__ float red[8];
    float* p = w + (size_t)blockIdx.x * SEQ;
    const int t = threadIdx.x;

    float v[8]; float m = -3.4e38f;
    #pragma unroll
    for (int i = 0; i < 8; ++i) { v[i] = p[t + i * 256]; m = fmaxf(m, v[i]); }
    #pragma unroll
    for (int o = 16; o; o >>= 1) m = fmaxf(m, __shfl_xor(m, o));
    if ((t & 31) == 0) red[t >> 5] = m;
    __syncthreads();
    m = red[0];
    #pragma unroll
    for (int i = 1; i < 8; ++i) m = fmaxf(m, red[i]);
    __syncthreads();

    float s = 0.f;
    #pragma unroll
    for (int i = 0; i < 8; ++i) { v[i] = __expf(v[i] - m); s += v[i]; }
    #pragma unroll
    for (int o = 16; o; o >>= 1) s += __shfl_xor(s, o);
    if ((t & 31) == 0) red[t >> 5] = s;
    __syncthreads();
    s = 0.f;
    #pragma unroll
    for (int i = 0; i < 8; ++i) s += red[i];

    const float inv = 1.f / s;
    #pragma unroll
    for (int i = 0; i < 8; ++i) p[t + i * 256] = v[i] * inv;
}

// ---------------------------------------------------------------------------
// LayerNorm over 1024-wide rows of (resid + y); optional fp32 / bf16 outputs.
// ---------------------------------------------------------------------------
__global__ __launch_bounds__(256) void layernorm1024(
    const float* __restrict__ resid, const float* __restrict__ y,
    const float* __restrict__ g, const float* __restrict__ b,
    float* __restrict__ outF, bf16* __restrict__ outB)
{
    __shared__ float rs[8], rs2[8];
    const size_t row = blockIdx.x;
    const float* rp = resid + row * DMODEL;
    const float* yp = y     + row * DMODEL;
    const int t = threadIdx.x;

    float v[4]; float s = 0.f, s2 = 0.f;
    #pragma unroll
    for (int i = 0; i < 4; ++i) {
        int c = t + i * 256;
        v[i] = rp[c] + yp[c];
        s += v[i]; s2 += v[i] * v[i];
    }
    #pragma unroll
    for (int o = 16; o; o >>= 1) { s += __shfl_xor(s, o); s2 += __shfl_xor(s2, o); }
    if ((t & 31) == 0) { rs[t >> 5] = s; rs2[t >> 5] = s2; }
    __syncthreads();
    s = 0.f; s2 = 0.f;
    #pragma unroll
    for (int i = 0; i < 8; ++i) { s += rs[i]; s2 += rs2[i]; }

    const float mu   = s * (1.f / DMODEL);
    const float var  = s2 * (1.f / DMODEL) - mu * mu;
    const float rinv = rsqrtf(var + 1e-6f);
    #pragma unroll
    for (int i = 0; i < 4; ++i) {
        int c = t + i * 256;
        float o = (v[i] - mu) * rinv * g[c] + b[c];
        if (outF) outF[row * DMODEL + c] = o;
        if (outB) outB[row * DMODEL + c] = f2bf(o);
    }
}

// ---------------------------------------------------------------------------
// fp32 -> bf16 cast (vectorized, n divisible by 4).
// ---------------------------------------------------------------------------
__global__ __launch_bounds__(256) void cast_f32_bf16(
    const float* __restrict__ in, bf16* __restrict__ out, size_t n)
{
    size_t i4 = (size_t)blockIdx.x * 256 + threadIdx.x;
    const size_t stride = (size_t)gridDim.x * 256;
    const size_t n4 = n >> 2;
    for (; i4 < n4; i4 += stride) {
        float4 f = reinterpret_cast<const float4*>(in)[i4];
        union { bf16 h[4]; uint2 u; } pk;
        pk.h[0] = f2bf(f.x); pk.h[1] = f2bf(f.y);
        pk.h[2] = f2bf(f.z); pk.h[3] = f2bf(f.w);
        reinterpret_cast<uint2*>(out)[i4] = pk.u;
    }
}

// ---------------------------------------------------------------------------
// fp32 (R x C) -> bf16 transposed (C x R), 32x32 LDS tiles.
// ---------------------------------------------------------------------------
__global__ __launch_bounds__(256) void cast_transpose_f32_bf16(
    const float* __restrict__ in, bf16* __restrict__ out, int R, int C)
{
    __shared__ float t[32][33];
    const int c0 = blockIdx.x * 32;
    const int r0 = blockIdx.y * 32;
    const int tid = threadIdx.x;
    const int r  = tid >> 3;
    const int c4 = (tid & 7) * 4;

    float4 f = *reinterpret_cast<const float4*>(in + (size_t)(r0 + r) * C + c0 + c4);
    t[r][c4 + 0] = f.x; t[r][c4 + 1] = f.y;
    t[r][c4 + 2] = f.z; t[r][c4 + 3] = f.w;
    __syncthreads();

    union { bf16 h[4]; uint2 u; } pk;
    #pragma unroll
    for (int i = 0; i < 4; ++i) pk.h[i] = f2bf(t[c4 + i][r]);
    *reinterpret_cast<uint2*>(out + (size_t)(c0 + r) * R + r0 + c4) = pk.u;
}

// ---------------------------------------------------------------------------
// Host orchestration
// ---------------------------------------------------------------------------
extern "C" void kernel_launch(void* const* d_in, const int* in_sizes, int n_in,
                              void* d_out, int out_size, void* d_ws, size_t ws_size,
                              hipStream_t stream)
{
    (void)in_sizes; (void)n_in; (void)out_size; (void)ws_size;

    const float* x       = (const float*)d_in[0];
    const float* enc     = (const float*)d_in[1];
    const float* padmask = (const float*)d_in[3];   // (B, S_ENC)

    const float* Wf[10] = {
        (const float*)d_in[4],  (const float*)d_in[6],
        (const float*)d_in[8],  (const float*)d_in[10],
        (const float*)d_in[12], (const float*)d_in[14],
        (const float*)d_in[16], (const float*)d_in[18],
        (const float*)d_in[20], (const float*)d_in[22] };
    const float* bias_[10] = {
        (const float*)d_in[5],  (const float*)d_in[7],
        (const float*)d_in[9],  (const float*)d_in[11],
        (const float*)d_in[13], (const float*)d_in[15],
        (const float*)d_in[17], (const float*)d_in[19],
        (const float*)d_in[21], (const float*)d_in[23] };
    const float* g1 = (const float*)d_in[24]; const float* be1 = (const float*)d_in[25];
    const float* g2 = (const float*)d_in[26]; const float* be2 = (const float*)d_in[27];
    const float* g3 = (const float*)d_in[28]; const float* be3 = (const float*)d_in[29];

    // d_out regions: out3 | w1 | w2
    float* out3 = (float*)d_out;
    float* w1   = out3 + (size_t)BATCH * SEQ * DMODEL;
    float* w2   = w1   + (size_t)BATCH * NHEADS * SEQ * SEQ;

    // ---- workspace arena ----
    char* p = (char*)d_ws;
    auto alloc = [&](size_t bytes) -> void* {
        void* r = (void*)p;
        p += (bytes + 255) & ~(size_t)255;
        return r;
    };
    const size_t MT = (size_t)BATCH * SEQ;   // 4096 token rows

    bf16* xb    = (bf16*)alloc(MT * DMODEL * 2);
    bf16* encb  = (bf16*)alloc(MT * DMODEL * 2);
    bf16* WbT[10];
    const int wR[10] = { DMODEL,DMODEL,DMODEL,DMODEL,DMODEL,DMODEL,DMODEL,DMODEL,
                         DMODEL, DFF };
    const int wC[10] = { DMODEL,DMODEL,DMODEL,DMODEL,DMODEL,DMODEL,DMODEL,DMODEL,
                         DFF,    DMODEL };
    for (int i = 0; i < 10; ++i)
        WbT[i] = (bf16*)alloc((size_t)wR[i] * wC[i] * 2);
    bf16*  qp    = (bf16*)alloc(MT * DMODEL * 2);
    bf16*  kp    = (bf16*)alloc(MT * DMODEL * 2);
    bf16*  vpT   = (bf16*)alloc(MT * DMODEL * 2);   // DMODEL x (B*S)
    bf16*  attnb = (bf16*)alloc(MT * DMODEL * 2);
    float* sa    = (float*)alloc(MT * DMODEL * 4);
    bf16*  out1b = (bf16*)alloc(MT * DMODEL * 2);
    float* out2f = (float*)alloc(MT * DMODEL * 4);
    bf16*  out2b = (bf16*)alloc(MT * DMODEL * 2);
    bf16*  hb    = (bf16*)alloc(MT * DFF * 2);

    const dim3 blk(256);
    const dim3 gcast(2048);
    const dim3 gproj(DMODEL / BN, (int)(MT / BM), 1);          // (16, 32)
    const dim3 gff1 (DFF    / BN, (int)(MT / BM), 1);          // (64, 32)
    const dim3 gsc  (SEQ / BN, SEQ / BM, BATCH * NHEADS);      // (32, 16, 32)
    const dim3 gav  (1,        SEQ / BM, BATCH * NHEADS);      // (1, 16, 32)

    // ---- casts: activations plain, weights transposed to N x K ----
    cast_f32_bf16<<<gcast, blk, 0, stream>>>(x,   xb,   MT * DMODEL);
    cast_f32_bf16<<<gcast, blk, 0, stream>>>(enc, encb, MT * DMODEL);
    for (int i = 0; i < 10; ++i) {
        dim3 gt(wC[i] / 32, wR[i] / 32);
        cast_transpose_f32_bf16<<<gt, blk, 0, stream>>>(Wf[i], WbT[i], wR[i], wC[i]);
    }

    const float iscale = 0.125f;  // 1/sqrt(depth=64)

    // ================= MHA 1 (self-attention, causal) =================
    gemm_wmma<0,1,0,0,1,0,0><<<gproj, blk, 0, stream>>>(
        xb, DMODEL, WbT[0], DMODEL, qp, DMODEL, bias_[0], nullptr, 1.f, DMODEL);
    gemm_wmma<0,1,0,0,1,0,0><<<gproj, blk, 0, stream>>>(
        xb, DMODEL, WbT[1], DMODEL, kp, DMODEL, bias_[1], nullptr, 1.f, DMODEL);
    gemm_wmma<0,1,0,0,1,1,0><<<gproj, blk, 0, stream>>>(       // V stored transposed
        xb, DMODEL, WbT[2], DMODEL, vpT, (int)MT, bias_[2], nullptr, 1.f, DMODEL);

    gemm_wmma<0,0,0,1,0,0,1><<<gsc, blk, 0, stream>>>(         // scores -> w1
        qp, DMODEL, kp, DMODEL, w1, SEQ, nullptr, nullptr, iscale, 64);
    softmax2048<<<BATCH * NHEADS * SEQ, blk, 0, stream>>>(w1);
    gemm_wmma<1,0,0,0,1,0,2><<<gav, blk, 0, stream>>>(         // w1 @ V
        w1, SEQ, vpT, (int)MT, attnb, 64, nullptr, nullptr, 1.f, SEQ);
    gemm_wmma<0,1,0,0,0,0,0><<<gproj, blk, 0, stream>>>(       // out projection
        attnb, DMODEL, WbT[3], DMODEL, sa, DMODEL, bias_[3], nullptr, 1.f, DMODEL);
    layernorm1024<<<(int)MT, blk, 0, stream>>>(x, sa, g1, be1, nullptr, out1b);

    // ================= MHA 2 (cross-attention, padding mask) =================
    gemm_wmma<0,1,0,0,1,0,0><<<gproj, blk, 0, stream>>>(
        out1b, DMODEL, WbT[4], DMODEL, qp, DMODEL, bias_[4], nullptr, 1.f, DMODEL);
    gemm_wmma<0,1,0,0,1,0,0><<<gproj, blk, 0, stream>>>(
        encb, DMODEL, WbT[5], DMODEL, kp, DMODEL, bias_[5], nullptr, 1.f, DMODEL);
    gemm_wmma<0,1,0,0,1,1,0><<<gproj, blk, 0, stream>>>(
        encb, DMODEL, WbT[6], DMODEL, vpT, (int)MT, bias_[6], nullptr, 1.f, DMODEL);

    gemm_wmma<0,0,0,2,0,0,1><<<gsc, blk, 0, stream>>>(         // scores -> w2
        qp, DMODEL, kp, DMODEL, w2, SEQ, nullptr, padmask, iscale, 64);
    softmax2048<<<BATCH * NHEADS * SEQ, blk, 0, stream>>>(w2);
    gemm_wmma<1,0,0,0,1,0,2><<<gav, blk, 0, stream>>>(         // w2 @ V
        w2, SEQ, vpT, (int)MT, attnb, 64, nullptr, nullptr, 1.f, SEQ);
    gemm_wmma<0,1,0,0,0,0,0><<<gproj, blk, 0, stream>>>(
        attnb, DMODEL, WbT[7], DMODEL, sa, DMODEL, bias_[7], nullptr, 1.f, DMODEL);
    // source quirk: residual uses x, not out1
    layernorm1024<<<(int)MT, blk, 0, stream>>>(x, sa, g2, be2, out2f, out2b);

    // ================= FFN =================
    gemm_wmma<0,1,1,0,1,0,0><<<gff1, blk, 0, stream>>>(        // relu(out2 @ W1 + b1)
        out2b, DMODEL, WbT[8], DMODEL, hb, DFF, bias_[8], nullptr, 1.f, DMODEL);
    gemm_wmma<0,1,0,0,0,0,0><<<gproj, blk, 0, stream>>>(       // h @ W2 + b2
        hb, DFF, WbT[9], DFF, sa, DMODEL, bias_[9], nullptr, 1.f, DFF);
    layernorm1024<<<(int)MT, blk, 0, stream>>>(out2f, sa, g3, be3, out3, nullptr);
}
